// AttentionConv2d_82832739271172
// MI455X (gfx1250) — compile-verified, matrix operand order checked by reading, not run on hardware
//
#include <hip/hip_runtime.h>
#include <cstdint>
#include <cstddef>

#define B_   8
#define C_   256
#define N_   4096
#define CR_  32
#define LOG2E 1.44269504088896340736f

typedef __bf16 bf16_t;
typedef __attribute__((ext_vector_type(16))) __bf16 v16bf;
typedef __attribute__((ext_vector_type(8)))  __bf16 v8bf;
typedef __attribute__((ext_vector_type(8)))  float  v8f;
typedef __attribute__((ext_vector_type(4)))  int    v4i_t;

// ---------------------------------------------------------------------------
// Async global->LDS staging (CDNA5).  16 bytes per lane per instruction,
// tracked by ASYNCcnt.  Falls back to load+ds_store if the builtin is absent.
// The builtin takes v4i pointers in AS1 (global) / AS3 (LDS); generic "shared"
// pointers carry the LDS byte offset in their low 32 bits (ISA 10.2 aperture
// mapping), so the integer casts below are address-correct.
// ---------------------------------------------------------------------------
#if __has_builtin(__builtin_amdgcn_global_load_async_to_lds_b128)
#define ASYNC_LDS 1
#else
#define ASYNC_LDS 0
#endif

__device__ __forceinline__ void copy16_g2l(const bf16_t* g, bf16_t* l) {
#if ASYNC_LDS
  __builtin_amdgcn_global_load_async_to_lds_b128(
      (__attribute__((address_space(1))) v4i_t*)(uintptr_t)g,
      (__attribute__((address_space(3))) v4i_t*)(uintptr_t)l,
      0, 0);
#else
  *(v8bf*)l = *(const v8bf*)g;
#endif
}

__device__ __forceinline__ void wait_streaming() {
#if ASYNC_LDS
#if __has_builtin(__builtin_amdgcn_s_wait_asynccnt)
  __builtin_amdgcn_s_wait_asynccnt(0);
#else
  asm volatile("s_wait_asynccnt 0x0" ::: "memory");
#endif
#endif
}

// ---------------------------------------------------------------------------
// WMMA operand loader.
// For both the 16-bit A (16x32) and B (32x16) layouts (ISA 7.12.2), a lane
// holds 16 bf16 values: K = koff + {0..7} (+8 for lanes 16..31) in VGPR0..3
// and K = koff + {16..23} (+8) in VGPR4..7, for row/col = lane&15.
// `rowbase` must already be:  base + (lane&15)*stride + ((lane>>4)&1)*8
// so the tile is two 16-byte contiguous loads per lane (global or LDS).
// ---------------------------------------------------------------------------
__device__ __forceinline__ v16bf load_tile(const bf16_t* rowbase, int koff) {
  v8bf lo = *(const v8bf*)(rowbase + koff);
  v8bf hi = *(const v8bf*)(rowbase + koff + 16);
  return __builtin_shufflevector(lo, hi, 0,1,2,3,4,5,6,7,8,9,10,11,12,13,14,15);
}

__device__ __forceinline__ v8f wmma_bf16(v16bf a, v16bf b, v8f c) {
  // (neg_a, A, neg_b, B, c_mod, C, reuse_a, reuse_b)
  return __builtin_amdgcn_wmma_f32_16x16x32_bf16(false, a, false, b, (short)0, c,
                                                 false, false);
}

// ---------------------------------------------------------------------------
// Kernel 1: convert weights fp32 -> bf16 (natural [o][c] layouts).
// ---------------------------------------------------------------------------
__global__ __launch_bounds__(256)
void convert_weights_kernel(const float* __restrict__ Wq,
                            const float* __restrict__ Wk,
                            const float* __restrict__ Wv,
                            bf16_t* __restrict__ wq,
                            bf16_t* __restrict__ wk,
                            bf16_t* __restrict__ wv) {
  int i = blockIdx.x * 256 + threadIdx.x;
  if (i < CR_ * C_) { wq[i] = (bf16_t)Wq[i]; wk[i] = (bf16_t)Wk[i]; }
  if (i < C_ * C_)  { wv[i] = (bf16_t)Wv[i]; }
}

// ---------------------------------------------------------------------------
// Kernel 2: x [B][C][N] fp32  ->  xT [B][N][C] bf16 (LDS tiled transpose).
// ---------------------------------------------------------------------------
__global__ __launch_bounds__(256)
void transpose_x_kernel(const float* __restrict__ x, bf16_t* __restrict__ xT) {
  __shared__ float tile[32][33];
  const int b  = blockIdx.z;
  const int c0 = blockIdx.x * 32;
  const int n0 = blockIdx.y * 32;
  const int tx = threadIdx.x;   // 0..31
  const int ty = threadIdx.y;   // 0..7
#pragma unroll
  for (int k = 0; k < 4; ++k) {
    int c = c0 + ty + k * 8;
    tile[ty + k * 8][tx] = x[((size_t)b * C_ + c) * N_ + n0 + tx];
  }
  __syncthreads();
#pragma unroll
  for (int k = 0; k < 4; ++k) {
    int n = ty + k * 8;
    xT[((size_t)b * N_ + n0 + n) * C_ + c0 + tx] = (bf16_t)tile[tx][n];
  }
}

// ---------------------------------------------------------------------------
// Kernel 3: q/k projections.  qT[b][n][o] = sum_c Wq[o][c]*xT[b][n][c] + bq[o]
// D = A(xT rows n, K=c) x B(Wq rows o, K=c contiguous).  D: m=n, lane=o.
// ---------------------------------------------------------------------------
__global__ __launch_bounds__(128)
void proj_qk_kernel(const bf16_t* __restrict__ xT,
                    const bf16_t* __restrict__ wq,
                    const bf16_t* __restrict__ wk,
                    const float*  __restrict__ bq,
                    const float*  __restrict__ bk,
                    bf16_t* __restrict__ qT,
                    bf16_t* __restrict__ kT) {
  const int lane = threadIdx.x & 31;
  const int wave = threadIdx.x >> 5;
  const int half = lane >> 4;
  const int jl   = lane & 15;
  const int ih   = half * 8;
  const int b    = blockIdx.x >> 6;                    // / (N/64)
  const int n0   = (blockIdx.x & 63) * 64 + wave * 16;

  const bf16_t* xrow   = xT + ((size_t)b * N_ + n0 + jl) * C_ + ih;
  const bf16_t* wqrow0 = wq + (size_t)(jl)      * C_ + ih;
  const bf16_t* wqrow1 = wq + (size_t)(16 + jl) * C_ + ih;
  const bf16_t* wkrow0 = wk + (size_t)(jl)      * C_ + ih;
  const bf16_t* wkrow1 = wk + (size_t)(16 + jl) * C_ + ih;

  v8f aq0 = {}, aq1 = {}, ak0 = {}, ak1 = {};
#pragma unroll
  for (int c0 = 0; c0 < C_; c0 += 32) {
    v16bf ax = load_tile(xrow, c0);
    aq0 = wmma_bf16(ax, load_tile(wqrow0, c0), aq0);
    aq1 = wmma_bf16(ax, load_tile(wqrow1, c0), aq1);
    ak0 = wmma_bf16(ax, load_tile(wkrow0, c0), ak0);
    ak1 = wmma_bf16(ax, load_tile(wkrow1, c0), ak1);
  }
  const float bq0 = bq[jl], bq1 = bq[16 + jl];
  const float bk0 = bk[jl], bk1 = bk[16 + jl];
#pragma unroll
  for (int r = 0; r < 8; ++r) {
    int n = n0 + ih + r;                  // D row m -> n
    size_t base = ((size_t)b * N_ + n) * CR_;
    qT[base + jl]      = (bf16_t)(aq0[r] + bq0);
    qT[base + 16 + jl] = (bf16_t)(aq1[r] + bq1);
    kT[base + jl]      = (bf16_t)(ak0[r] + bk0);
    kT[base + 16 + jl] = (bf16_t)(ak1[r] + bk1);
  }
}

// ---------------------------------------------------------------------------
// Kernel 4: v projection.  v[b][c][n] = sum_ci Wv[c][ci]*xT[b][n][ci] + bv[c]
// D = A(Wv rows c, K=ci) x B(xT rows n, K=ci).  D: m=c, lane=n.
// ---------------------------------------------------------------------------
__global__ __launch_bounds__(128)
void proj_v_kernel(const bf16_t* __restrict__ xT,
                   const bf16_t* __restrict__ wv,
                   const float*  __restrict__ bv,
                   bf16_t* __restrict__ vM) {
  const int lane = threadIdx.x & 31;
  const int wave = threadIdx.x >> 5;
  const int half = lane >> 4;
  const int jl   = lane & 15;
  const int ih   = half * 8;
  const int bx   = blockIdx.x;
  const int b    = bx / (16 * 64);
  const int rem  = bx % (16 * 64);
  const int c0   = (rem / 64) * 16;
  const int n0   = (rem % 64) * 64 + wave * 16;

  const bf16_t* wrow = wv + ((size_t)(c0 + jl)) * C_ + ih;
  const bf16_t* xrow = xT + ((size_t)b * N_ + n0 + jl) * C_ + ih;

  v8f acc = {};
#pragma unroll
  for (int ci = 0; ci < C_; ci += 32)
    acc = wmma_bf16(load_tile(wrow, ci), load_tile(xrow, ci), acc);

#pragma unroll
  for (int r = 0; r < 8; ++r) {
    int c = c0 + ih + r;                  // D row m -> c
    vM[((size_t)b * C_ + c) * N_ + n0 + jl] = (bf16_t)(acc[r] + bv[c]);
  }
}

// ---------------------------------------------------------------------------
// Kernel 5: flash attention + gamma scale + residual.
// Block = 4 waves = 64 j-columns, one batch.  The Q rows and V tiles for an
// i-step are IDENTICAL across the 4 waves, so they are staged once per block
// into double-buffered LDS with async global->LDS copies, overlapped with the
// WMMA work on the current buffer.  Per wave per 32-i step:
//   S(i,j)  = A(lds_q rows i, K=32) x B(kT rows j, K=32)        [2 wmma]
//   O(j,c) += A(P rows j via LDS,  K=i) x B(lds_v rows c, K=i)  [16 wmma]
// ---------------------------------------------------------------------------
__device__ __forceinline__ void stage_tiles(const bf16_t* __restrict__ qTb,
                                            const bf16_t* __restrict__ vb,
                                            bf16_t* lv, bf16_t* lq,
                                            int i0, int tid) {
  // V block: 256 c x 32 i bf16 (16 KB), LDS layout [c][i_local], 64B rows.
#pragma unroll
  for (int s = 0; s < 8; ++s) {
    int e  = tid + s * 128;
    int c  = e >> 2;
    int ic = (e & 3) * 8;
    copy16_g2l(vb + (size_t)c * N_ + i0 + ic, lv + c * 32 + ic);
  }
  // Q block: 32 rows x 32 o bf16 (2 KB), LDS layout [i_local][o].
  {
    int row = tid >> 2;
    int col = (tid & 3) * 8;
    copy16_g2l(qTb + (size_t)(i0 + row) * CR_ + col, lq + row * 32 + col);
  }
}

__global__ __launch_bounds__(128)
void flash_attn_kernel(const bf16_t* __restrict__ qT,
                       const bf16_t* __restrict__ kT,
                       const bf16_t* __restrict__ vM,
                       const float*  __restrict__ x,
                       const float*  __restrict__ gamma,
                       float* __restrict__ out) {
  __shared__ __align__(16) bf16_t lds_v[2][C_ * 32];   // 2 x 16 KB
  __shared__ __align__(16) bf16_t lds_q[2][32 * CR_];  // 2 x  2 KB
  __shared__ __align__(16) bf16_t lds_p[4][16 * 32];   // per-wave P tiles
  __shared__ float sc_lds[4][16];
  __shared__ float l_lds[4][16];

  const int tid  = threadIdx.x;
  const int lane = tid & 31;
  const int wave = tid >> 5;
  const int half = lane >> 4;
  const int jl   = lane & 15;
  const int ih   = half * 8;
  const int b    = blockIdx.x >> 6;
  const int j0   = (blockIdx.x & 63) * 64 + wave * 16;

  const bf16_t* qTb = qT + (size_t)b * N_ * CR_;
  const bf16_t* kTb = kT + (size_t)b * N_ * CR_;
  const bf16_t* vb  = vM + (size_t)b * C_ * N_;

  // k columns for this wave: fixed B operand of the first GEMM.
  const v16bf b_k = load_tile(kTb + (size_t)(j0 + jl) * CR_ + ih, 0);

  v8f acc[16] = {};
  float m_run = -3.0e38f;
  float l_run = 0.0f;

  // Prime the pipeline.
  stage_tiles(qTb, vb, lds_v[0], lds_q[0], 0, tid);
  wait_streaming();
  __syncthreads();

  int cur = 0;
  for (int i0 = 0; i0 < N_; i0 += 32) {
    // Kick off the async fill of the other buffer while we compute.
    if (i0 + 32 < N_)
      stage_tiles(qTb, vb, lds_v[cur ^ 1], lds_q[cur ^ 1], i0 + 32, tid);

    // ---- S = q^T k for i0..i0+31 (A operand from LDS) ----
    v16bf a0 = load_tile(&lds_q[cur][jl * 32 + ih], 0);
    v16bf a1 = load_tile(&lds_q[cur][(16 + jl) * 32 + ih], 0);
    v8f z = {};
    v8f s0 = wmma_bf16(a0, b_k, z);   // lanes: n=j, VGPR r: i_local = ih+r
    v8f s1 = wmma_bf16(a1, b_k, z);   // i_local = 16+ih+r

    // ---- online softmax over i (per column j) ----
    float tmax = -3.0e38f;
#pragma unroll
    for (int r = 0; r < 8; ++r) {
      tmax = fmaxf(tmax, s0[r]);
      tmax = fmaxf(tmax, s1[r]);
    }
    tmax = fmaxf(tmax, __shfl_xor(tmax, 16, 32));
    const float m_new = fmaxf(m_run, tmax);
    const float corr  = exp2f((m_run - m_new) * LOG2E);

    float p0[8], p1[8], lsum = 0.0f;
#pragma unroll
    for (int r = 0; r < 8; ++r) {
      p0[r] = exp2f((s0[r] - m_new) * LOG2E);
      p1[r] = exp2f((s1[r] - m_new) * LOG2E);
      lsum += p0[r] + p1[r];
    }
    lsum += __shfl_xor(lsum, 16, 32);
    l_run = l_run * corr + lsum;
    m_run = m_new;

    // ---- stage P (bf16) in per-wave LDS as [j][i_local] (A-layout reload).
    // Same-wave DS ops are in order, so no block barrier is needed here.
#pragma unroll
    for (int r = 0; r < 8; ++r) {
      lds_p[wave][jl * 32 + ih + r]      = (bf16_t)p0[r];
      lds_p[wave][jl * 32 + 16 + ih + r] = (bf16_t)p1[r];
    }
    sc_lds[wave][jl] = corr;

    const v16bf a_p = load_tile(&lds_p[wave][jl * 32 + ih], 0);
    float sr[8];
#pragma unroll
    for (int r = 0; r < 8; ++r) sr[r] = sc_lds[wave][ih + r];

    // rescale running output, then accumulate P x V^T over all 256 channels
#pragma unroll
    for (int t = 0; t < 16; ++t) {
#pragma unroll
      for (int r = 0; r < 8; ++r) acc[t][r] *= sr[r];
    }
#pragma unroll
    for (int t = 0; t < 16; ++t) {
      v16bf b_v = load_tile(&lds_v[cur][(t * 16 + jl) * 32 + ih], 0);
      acc[t] = wmma_bf16(a_p, b_v, acc[t]);
    }

    // Buffer rotation: all async fills done + all waves past their reads.
    wait_streaming();
    __syncthreads();
    cur ^= 1;
  }

  // ---- epilogue: gamma / l, residual add, contiguous 32B stores per lane ----
  l_lds[wave][jl] = l_run;
  __syncthreads();
  const float g = gamma[0];
  float invr[8];
#pragma unroll
  for (int r = 0; r < 8; ++r) invr[r] = g / l_lds[wave][ih + r];

#pragma unroll
  for (int t = 0; t < 16; ++t) {
    int c = t * 16 + jl;                          // O: lane n -> channel
    size_t idx0 = ((size_t)b * C_ + c) * N_ + j0 + ih;  // VGPR r -> j (contig)
    v8f xr = *(const v8f*)&x[idx0];
    v8f o;
#pragma unroll
    for (int r = 0; r < 8; ++r) o[r] = acc[t][r] * invr[r] + xr[r];
    *(v8f*)&out[idx0] = o;
  }
}

// ---------------------------------------------------------------------------
// Workspace layout (bytes)
// ---------------------------------------------------------------------------
static const size_t OFF_XT = 0;                                      // 16 MiB
static const size_t OFF_QT = OFF_XT + (size_t)B_ * N_ * C_  * 2;     //  2 MiB
static const size_t OFF_KT = OFF_QT + (size_t)B_ * N_ * CR_ * 2;     //  2 MiB
static const size_t OFF_V  = OFF_KT + (size_t)B_ * N_ * CR_ * 2;     // 16 MiB
static const size_t OFF_WQ = OFF_V  + (size_t)B_ * C_ * N_  * 2;
static const size_t OFF_WK = OFF_WQ + (size_t)CR_ * C_ * 2;
static const size_t OFF_WV = OFF_WK + (size_t)CR_ * C_ * 2;

extern "C" void kernel_launch(void* const* d_in, const int* in_sizes, int n_in,
                              void* d_out, int out_size, void* d_ws, size_t ws_size,
                              hipStream_t stream) {
  const float* x     = (const float*)d_in[0];
  const float* Wq    = (const float*)d_in[1];
  const float* bq    = (const float*)d_in[2];
  const float* Wk    = (const float*)d_in[3];
  const float* bk    = (const float*)d_in[4];
  const float* Wv    = (const float*)d_in[5];
  const float* bv    = (const float*)d_in[6];
  const float* gamma = (const float*)d_in[7];
  float* out = (float*)d_out;

  char* ws = (char*)d_ws;
  bf16_t* xT = (bf16_t*)(ws + OFF_XT);
  bf16_t* qT = (bf16_t*)(ws + OFF_QT);
  bf16_t* kT = (bf16_t*)(ws + OFF_KT);
  bf16_t* vM = (bf16_t*)(ws + OFF_V);
  bf16_t* wq = (bf16_t*)(ws + OFF_WQ);
  bf16_t* wk = (bf16_t*)(ws + OFF_WK);
  bf16_t* wv = (bf16_t*)(ws + OFF_WV);

  convert_weights_kernel<<<dim3((C_ * C_ + 255) / 256), 256, 0, stream>>>(
      Wq, Wk, Wv, wq, wk, wv);

  transpose_x_kernel<<<dim3(C_ / 32, N_ / 32, B_), dim3(32, 8), 0, stream>>>(x, xT);

  proj_qk_kernel<<<dim3(B_ * (N_ / 64)), 128, 0, stream>>>(
      xT, wq, wk, bq, bk, qT, kT);

  proj_v_kernel<<<dim3(B_ * (C_ / 16) * (N_ / 64)), 128, 0, stream>>>(
      xT, wv, bv, vM);

  flash_attn_kernel<<<dim3(B_ * (N_ / 64)), 128, 0, stream>>>(
      qT, kT, vM, x, gamma, out);
}
